// XLDecoder_36507222016074
// MI455X (gfx1250) — matrix operand order repaired and checked by reading, not compiled, not run on hardware
//
#include <hip/hip_runtime.h>

#define Bsz 8
#define Ssz 128
#define Esz 512
#define Dsz 1024
#define Hsz 16
#define Lsz 4
#define Vsz 32000
#define DHsz 64
#define FFsz 4096
#define SCALEF 0.125f
#define NEGF (-1e30f)

typedef __attribute__((ext_vector_type(16))) _Float16 v16h;
typedef __attribute__((ext_vector_type(8)))  float    v8f;

// ---------------------------------------------------------------------------
// Batched strided WMMA GEMM with f16 B operand:
//   C = act( scale*(A x B) + bias + resid + mask )
// Invariants (all call sites): sAk==1, sCn==1, sRn==1, M%128==0, N%32==0,
// K%32==0, >=8B alignment everywhere.
// Workgroup = 256 threads = 8 waves -> 128x32 C macro-tile, wave = 16x32
// (two v_wmma_f32_16x16x32_f16 per K-step, A fragment reused for both).
// A: f32, direct global_load_b128 + packed cvt.  B: f16 (pre-converted once
// per pass), 32x32 slab staged to LDS in fragment-ready layout shared by all
// 8 waves.  BKC=1 (k-contiguous, weights): per-lane 8B async copy via
// global_load_async_to_lds_b64 (ASYNCcnt), overlapped with A loads.
// BKC=0 (n-contiguous, attention V): manual b64 load + ds_store_b16 scatter.
// ---------------------------------------------------------------------------
template<int BKC>
__global__ __launch_bounds__(256) void k_gemm(
    const float* __restrict__ A, const _Float16* __restrict__ Bm, float* __restrict__ C,
    int M, int N, int K,
    long sAm, long sBk, long sBn, long sCm,
    long bAo, long bAi, long bBo, long bBi, long bCo, long bCi, int hdiv,
    const float* __restrict__ bias,
    const float* __restrict__ resid, long sRm,
    const float* __restrict__ amask, long mRow,
    float scale, int act)
{
    __shared__ _Float16 lb[64][16] __attribute__((aligned(32)));   // 2 sub-tiles x 32 lanes x 16 halves

    const int tid  = threadIdx.x;
    const int wave = tid >> 5;
    const int lane = tid & 31;
    const int half = lane >> 4, row = lane & 15;
    const int tilesN = N >> 5;
    const int bm = blockIdx.x / tilesN;
    const int bn = blockIdx.x % tilesN;
    const int bt = blockIdx.y;
    const int bo = bt / hdiv, bi = bt % hdiv;
    const float*    Ab = A  + (long)bo * bAo + (long)bi * bAi;
    const _Float16* Bb = Bm + (long)bo * bBo + (long)bi * bBi;
    float*          Cb = C  + (long)bo * bCo + (long)bi * bCi;

    const long am = (long)bm * 128 + wave * 16 + row;    // A fragment: lane carries M
    const float* arow = Ab + am * sAm;

    // ---- staging geometry (loop-invariant) ----
    // BKC=1: thread t stages 4 k-contiguous halves of column sg_n.
    const int sg_n = tid >> 3;
    const int sg_k = (tid & 7) << 2;
    const int rowi = ((sg_n >> 4) << 5) + (sg_k & 16) + (sg_n & 15);
    const unsigned ldsoff = (unsigned)(unsigned long long)&lb[rowi][sg_k & 15];
    const _Float16* bcol = Bb + (long)(bn * 32 + sg_n) * sBn + sg_k;
    // BKC=0: thread t stages 4 n-contiguous halves of k-row sg_k2.
    const int sg_k2 = tid >> 3;
    const int sg_n2 = (tid & 7) << 2;
    const _Float16* brow = Bb + bn * 32 + sg_n2;

    v8f acc0 = {0.f,0.f,0.f,0.f,0.f,0.f,0.f,0.f};
    v8f acc1 = {0.f,0.f,0.f,0.f,0.f,0.f,0.f,0.f};

    for (int k0 = 0; k0 < K; k0 += 32) {
        // ---- stage B 32x32 f16 slab into LDS ----
        if (BKC) {
            // 8B per lane, fragment-ready destination; tracked by ASYNCcnt.
            asm volatile("global_load_async_to_lds_b64 %0, %1, off"
                         :: "v"(ldsoff), "v"(bcol + k0) : "memory");
        } else {
            _Float16 h[4];
            __builtin_memcpy(h, brow + (long)(k0 + sg_k2) * sBk, 8);
            const int e = sg_k2 & 15;
            const int lbase = sg_k2 & 16;
#pragma unroll
            for (int j = 0; j < 4; ++j) {
                const int nn = sg_n2 + j;
                lb[((nn >> 4) << 5) + lbase + (nn & 15)][e] = h[j];
            }
        }

        // ---- A fragment: 4x global_load_b128 + packed cvt (overlaps async B) ----
        if (k0 + 32 < K) __builtin_prefetch(arow + k0 + 32, 0, 1);  // global_prefetch_b8
        const float4 a0 = *(const float4*)(arow + k0 + (half << 4));
        const float4 a1 = *(const float4*)(arow + k0 + (half << 4) + 4);
        const float4 a2 = *(const float4*)(arow + k0 + (half << 4) + 8);
        const float4 a3 = *(const float4*)(arow + k0 + (half << 4) + 12);
        v16h a;
        a[0]  = (_Float16)a0.x; a[1]  = (_Float16)a0.y; a[2]  = (_Float16)a0.z; a[3]  = (_Float16)a0.w;
        a[4]  = (_Float16)a1.x; a[5]  = (_Float16)a1.y; a[6]  = (_Float16)a1.z; a[7]  = (_Float16)a1.w;
        a[8]  = (_Float16)a2.x; a[9]  = (_Float16)a2.y; a[10] = (_Float16)a2.z; a[11] = (_Float16)a2.w;
        a[12] = (_Float16)a3.x; a[13] = (_Float16)a3.y; a[14] = (_Float16)a3.z; a[15] = (_Float16)a3.w;

        if (BKC) asm volatile("s_wait_asynccnt 0" ::: "memory");
        __syncthreads();

        // ---- B fragments: contiguous 32B v16h per lane from LDS ----
        const v16h* bp = (const v16h*)lb;
        const v16h b0 = bp[lane];
        const v16h b1 = bp[32 + lane];

        acc0 = __builtin_amdgcn_wmma_f32_16x16x32_f16(false, a, false, b0, (short)0, acc0, false, false);
        acc1 = __builtin_amdgcn_wmma_f32_16x16x32_f16(false, a, false, b1, (short)0, acc1, false, false);
        __syncthreads();
    }

    // ---- fused epilogue ----
    const long cmBase = (long)bm * 128 + wave * 16;
#pragma unroll
    for (int g = 0; g < 2; ++g) {
        const v8f accv = g ? acc1 : acc0;
        const long cn = (long)bn * 32 + g * 16 + row;   // C: lane carries N
        const float bsv = bias  ? bias[cn] : 0.f;
        const float mkv = amask ? (1.0f - amask[(long)bo * mRow + cn]) * NEGF : 0.f;
#pragma unroll
        for (int r2 = 0; r2 < 8; ++r2) {
            const long cm = cmBase + r2 + (half << 3);  // C: VGPR r -> M = r + 8*half
            float v = accv[r2] * scale + bsv + mkv;
            if (resid) v += resid[cm * sRm + cn];
            if (act == 1) v = 0.5f * v * (1.0f + erff(v * 0.70710678118f));  // exact GELU
            Cb[cm * sCm + cn] = v;
        }
    }
}

// ---------------------------------------------------------------------------
// Support kernels
// ---------------------------------------------------------------------------
__global__ void k_cvt16(const float* __restrict__ in, _Float16* __restrict__ out, long n) {
    long i = ((long)blockIdx.x * blockDim.x + threadIdx.x) * 4;
    if (i >= n) return;
    const float4 v = *(const float4*)(in + i);
    out[i + 0] = (_Float16)v.x;
    out[i + 1] = (_Float16)v.y;
    out[i + 2] = (_Float16)v.z;
    out[i + 3] = (_Float16)v.w;
}

__global__ void k_embed(const int* __restrict__ ids, const float* __restrict__ emb,
                        float* __restrict__ core) {
    int sb = blockIdx.x; int s = sb / Bsz, b = sb % Bsz;
    int id = ids[b * Ssz + s];
    const float* src = emb + (long)id * Dsz;
    float* dst = core + (long)sb * Dsz;
    for (int d = threadIdx.x; d < Dsz; d += blockDim.x) dst[d] = src[d];
}

__global__ void k_posr(float* __restrict__ r) {
    int i = blockIdx.x;
    float pos = (float)(Ssz - 1 - i);
    for (int c = threadIdx.x; c < Dsz; c += blockDim.x) {
        int f = (c < Dsz / 2) ? c : c - Dsz / 2;
        float invf = powf(10000.0f, -(2.0f * (float)f) / (float)Dsz);
        float ang = pos * invf;
        r[(long)i * Dsz + c] = (c < Dsz / 2) ? sinf(ang) : cosf(ang);
    }
}

// [A1,A2,D] -> [A2,A1,D]
__global__ void k_transpose(const float* __restrict__ in, float* __restrict__ out,
                            int A1, int A2) {
    long idx = (long)blockIdx.x * blockDim.x + threadIdx.x;
    long total = (long)A1 * A2 * Dsz;
    if (idx >= total) return;
    int d = idx % Dsz; long r = idx / Dsz;
    int a2 = r % A2; long a1 = r / A2;
    out[((long)a2 * A1 + a1) * Dsz + d] = in[idx];
}

__global__ void k_qbias(const float* __restrict__ heads, const float* __restrict__ rw,
                        const float* __restrict__ rr, float* __restrict__ qrw,
                        float* __restrict__ qrr) {
    long idx = (long)blockIdx.x * blockDim.x + threadIdx.x;
    if (idx >= (long)Ssz * Bsz * Dsz) return;
    int d = idx % Dsz; long sb = idx / Dsz;
    float q = heads[sb * 3 * Dsz + d];
    qrw[idx] = q + rw[d];
    qrr[idx] = q + rr[d];
}

// score[bt,i,j] = j<=i ? (AC + BDraw[bt,i,S-1-i+j]) * SCALE : NEG   (rel_shift+causal)
__global__ void k_shift_mask(float* __restrict__ sc, const float* __restrict__ bd) {
    long idx = (long)blockIdx.x * blockDim.x + threadIdx.x;
    if (idx >= (long)Bsz * Hsz * Ssz * Ssz) return;
    int j = idx % Ssz; long r = idx / Ssz; int i = r % Ssz; long bt = r / Ssz;
    float v;
    if (j <= i) v = (sc[idx] + bd[(bt * Ssz + i) * Ssz + (Ssz - 1 - i + j)]) * SCALEF;
    else        v = NEGF;
    sc[idx] = v;
}

__global__ __launch_bounds__(256) void k_softmax(float* __restrict__ x, int cols) {
    long row = blockIdx.x;
    float* p = x + row * (long)cols;
    __shared__ float red[256];
    int tid = threadIdx.x;
    float mx = -3.4e38f;
    for (int c = tid; c < cols; c += 256) mx = fmaxf(mx, p[c]);
    red[tid] = mx; __syncthreads();
    for (int o = 128; o > 0; o >>= 1) { if (tid < o) red[tid] = fmaxf(red[tid], red[tid+o]); __syncthreads(); }
    mx = red[0]; __syncthreads();
    float s = 0.f;
    for (int c = tid; c < cols; c += 256) s += expf(p[c] - mx);
    red[tid] = s; __syncthreads();
    for (int o = 128; o > 0; o >>= 1) { if (tid < o) red[tid] += red[tid+o]; __syncthreads(); }
    float inv = 1.0f / red[0];
    for (int c = tid; c < cols; c += 256) p[c] = expf(p[c] - mx) * inv;
}

__global__ __launch_bounds__(256) void k_layernorm(const float* __restrict__ x,
                                                   const float* __restrict__ g,
                                                   const float* __restrict__ bb,
                                                   float* __restrict__ y) {
    long row = blockIdx.x;
    const float* p = x + row * Dsz;
    __shared__ float rs[256], rq[256];
    int tid = threadIdx.x;
    float s = 0.f, q = 0.f;
    for (int c = tid; c < Dsz; c += 256) { float v = p[c]; s += v; q += v * v; }
    rs[tid] = s; rq[tid] = q; __syncthreads();
    for (int o = 128; o > 0; o >>= 1) { if (tid < o) { rs[tid] += rs[tid+o]; rq[tid] += rq[tid+o]; } __syncthreads(); }
    float mean = rs[0] / (float)Dsz;
    float var  = rq[0] / (float)Dsz - mean * mean;
    float inv  = rsqrtf(var + 1e-5f);
    for (int c = tid; c < Dsz; c += 256) y[row * Dsz + c] = (p[c] - mean) * inv * g[c] + bb[c];
}

__global__ __launch_bounds__(256) void k_mode(const float* __restrict__ x,
                                              const float* __restrict__ w,
                                              const float* __restrict__ b,
                                              float* __restrict__ out) {
    long row = blockIdx.x;
    __shared__ float rs[256];
    int tid = threadIdx.x;
    float s = 0.f;
    for (int c = tid; c < Dsz; c += 256) s += x[row * Dsz + c] * w[c];
    rs[tid] = s; __syncthreads();
    for (int o = 128; o > 0; o >>= 1) { if (tid < o) rs[tid] += rs[tid+o]; __syncthreads(); }
    if (tid == 0) out[row] = 1.0f / (1.0f + expf(-(rs[0] + b[0])));
}

// vocab_prob = vocab_prob*mode; then scatter-add (1-mode)*input_prob at input_ids
__global__ __launch_bounds__(256) void k_mix(float* __restrict__ vocab,
                                             const float* __restrict__ mode,
                                             const float* __restrict__ iprob,
                                             const int* __restrict__ ids) {
    int bs = blockIdx.x; int b = bs / Ssz;
    float m = mode[bs];
    float* vp = vocab + (long)bs * Vsz;
    for (int c = threadIdx.x; c < Vsz; c += 256) vp[c] *= m;
    __syncthreads();
    const float* ip = iprob + (long)bs * Esz;
    float om = 1.0f - m;
    for (int j = threadIdx.x; j < Esz; j += 256)
        atomicAdd(&vp[ids[b * Esz + j]], om * ip[j]);
}

__global__ void k_zero2(float* a) { if (threadIdx.x < 2) a[threadIdx.x] = 0.f; }

__global__ void k_loss(const float* __restrict__ vocab, const int* __restrict__ tgt,
                       float* __restrict__ accum) {
    int t = blockIdx.x * blockDim.x + threadIdx.x;
    if (t >= Bsz * Ssz) return;
    int y = tgt[t];
    float pr = vocab[(long)t * Vsz + y];
    float valid = (y != 0) ? 1.0f : 0.0f;
    atomicAdd(&accum[0], -logf(pr + 1e-6f) * valid);
    atomicAdd(&accum[1], valid);
}

__global__ void k_finish(const float* __restrict__ accum, float* __restrict__ out) {
    if (threadIdx.x < Bsz) out[threadIdx.x] = accum[0] / accum[1];
}

// ---------------------------------------------------------------------------
// Host side
// ---------------------------------------------------------------------------
static inline void cvt16(hipStream_t st, const float* src, _Float16* dst, long n) {
    long blocks = (n / 4 + 255) / 256;
    k_cvt16<<<dim3((unsigned)blocks), dim3(256), 0, st>>>(src, dst, n);
}

static inline void gemm(hipStream_t st,
                        const float* A, const _Float16* Bm, float* C,
                        int M, int N, int K,
                        long sAm, long sBk, long sBn, long sCm,
                        int batch = 1, int hdiv = 1,
                        long bAo = 0, long bAi = 0, long bBo = 0, long bBi = 0,
                        long bCo = 0, long bCi = 0,
                        const float* bias = nullptr,
                        const float* resid = nullptr, long sRm = 0,
                        const float* amask = nullptr, long mRow = 0,
                        float scale = 1.0f, int act = 0)
{
    dim3 grid((M / 128) * (N / 32), batch, 1);
    if (sBk == 1)
        k_gemm<1><<<grid, dim3(256), 0, st>>>(A, Bm, C, M, N, K, sAm, sBk, sBn, sCm,
            bAo, bAi, bBo, bBi, bCo, bCi, hdiv, bias, resid, sRm, amask, mRow, scale, act);
    else
        k_gemm<0><<<grid, dim3(256), 0, st>>>(A, Bm, C, M, N, K, sAm, sBk, sBn, sCm,
            bAo, bAi, bBo, bBi, bCo, bCi, hdiv, bias, resid, sRm, amask, mRow, scale, act);
}

extern "C" void kernel_launch(void* const* d_in, const int* in_sizes, int n_in,
                              void* d_out, int out_size, void* d_ws, size_t ws_size,
                              hipStream_t stream) {
    (void)in_sizes; (void)n_in; (void)out_size; (void)ws_size;
    const int*   input_ids   = (const int*)d_in[0];
    const float* encoder_rep = (const float*)d_in[1];
    const float* input_mask  = (const float*)d_in[2];
    const int*   decode_in   = (const int*)d_in[3];
    const int*   decode_tgt  = (const int*)d_in[4];
    const float* word_emb    = (const float*)d_in[5];
    const float* qkv_w = (const float*)d_in[6];
    const float* r_w   = (const float*)d_in[7];
    const float* o_w   = (const float*)d_in[8];
    const float* kv_w  = (const float*)d_in[9];
    const float* q_w   = (const float*)d_in[10];
    const float* io_w  = (const float*)d_in[11];
    const float* rr_b  = (const float*)d_in[12];
    const float* rw_b  = (const float*)d_in[13];
    const float* ln1g  = (const float*)d_in[14];
    const float* ln1b  = (const float*)d_in[15];
    const float* ln2g  = (const float*)d_in[16];
    const float* ln2b  = (const float*)d_in[17];
    const float* f1w   = (const float*)d_in[18];
    const float* f1b   = (const float*)d_in[19];
    const float* f2w   = (const float*)d_in[20];
    const float* f2b   = (const float*)d_in[21];
    const float* ln3g  = (const float*)d_in[22];
    const float* ln3b  = (const float*)d_in[23];
    const float* out_w = (const float*)d_in[24];
    const float* out_b = (const float*)d_in[25];
    const float* cp_w  = (const float*)d_in[26];
    const float* cp_b  = (const float*)d_in[27];
    const float* md_w  = (const float*)d_in[28];
    const float* md_b  = (const float*)d_in[29];

    const int SB = Ssz * Bsz;          // 1024 rows (also == B*S)
    char* wsp = (char*)d_ws; size_t off = 0;
    auto allocF = [&](size_t elems) -> float* {
        float* p = (float*)(wsp + off);
        off += ((elems * 4 + 255) / 256) * 256;
        return p;
    };
    auto allocH = [&](size_t elems) -> _Float16* {
        _Float16* p = (_Float16*)(wsp + off);
        off += ((elems * 2 + 255) / 256) * 256;
        return p;
    };
    float* rbuf  = allocF((size_t)Ssz * Dsz);
    float* rk    = allocF((size_t)Ssz * Dsz);
    float* core  = allocF((size_t)SB * Dsz);
    float* enc_t = allocF((size_t)Esz * Bsz * Dsz);
    float* heads = allocF((size_t)SB * 3 * Dsz);
    float* qrw   = allocF((size_t)SB * Dsz);
    float* qrr   = allocF((size_t)SB * Dsz);
    float* scS   = allocF((size_t)Bsz * Hsz * Ssz * Ssz);
    float* bdr   = allocF((size_t)Bsz * Hsz * Ssz * Ssz);
    float* vec   = allocF((size_t)SB * Dsz);
    float* xbuf  = allocF((size_t)SB * Dsz);
    float* xln   = allocF((size_t)SB * Dsz);
    float* q2    = allocF((size_t)SB * Dsz);
    float* kv    = allocF((size_t)Esz * Bsz * 2 * Dsz);
    float* sc2   = allocF((size_t)Bsz * Hsz * Ssz * Esz);
    float* vec2  = allocF((size_t)SB * Dsz);
    float* ybuf  = allocF((size_t)SB * Dsz);
    float* yln   = allocF((size_t)SB * Dsz);
    float* hff   = allocF((size_t)SB * FFsz);
    float* cbuf  = allocF((size_t)SB * Dsz);
    float* cout  = allocF((size_t)SB * Dsz);
    float* obuf  = allocF((size_t)SB * Dsz);
    float* vocab = allocF((size_t)SB * Vsz);
    float* cph   = allocF((size_t)SB * Dsz);
    float* ilog  = allocF((size_t)SB * Esz);
    float* modeb = allocF((size_t)SB);
    float* accum = allocF(2);
    // f16 operand pool
    _Float16* word16  = allocH((size_t)Vsz * Dsz);
    _Float16* enc16   = allocH((size_t)Bsz * Esz * Dsz);
    _Float16* wscr    = allocH((size_t)FFsz * Dsz);      // per-weight scratch (max 4M elems)
    _Float16* heads16 = allocH((size_t)SB * 3 * Dsz);
    _Float16* kv16    = allocH((size_t)Esz * Bsz * 2 * Dsz);
    _Float16* rk16    = allocH((size_t)Ssz * Dsz);

    // One-time f16 conversions for pass-invariant B operands
    cvt16(stream, word_emb, word16, (long)Vsz * Dsz);
    cvt16(stream, encoder_rep, enc16, (long)Bsz * Esz * Dsz);

    // Embedding gather, positional encoding, encoder transpose [B,E,D]->[E,B,D]
    k_embed<<<SB, 256, 0, stream>>>(decode_in, word_emb, core);
    k_posr<<<Ssz, 256, 0, stream>>>(rbuf);
    { long tot = (long)Bsz * Esz * Dsz;
      k_transpose<<<(tot + 255) / 256, 256, 0, stream>>>(encoder_rep, enc_t, Bsz, Esz); }

    for (int l = 0; l < Lsz; ++l) {
        const float* Wqkv = qkv_w + (size_t)l * 3 * Dsz * Dsz;
        const float* Wr   = r_w   + (size_t)l * Dsz * Dsz;
        const float* Wo   = o_w   + (size_t)l * Dsz * Dsz;
        const float* Wkv  = kv_w  + (size_t)l * 2 * Dsz * Dsz;
        const float* Wq   = q_w   + (size_t)l * Dsz * Dsz;
        const float* Wio  = io_w  + (size_t)l * Dsz * Dsz;
        const float* Wf1  = f1w   + (size_t)l * FFsz * Dsz;
        const float* Wf2  = f2w   + (size_t)l * Dsz * FFsz;

        // heads = core @ qkv_w^T  [SB x 3D]
        cvt16(stream, Wqkv, wscr, 3L * Dsz * Dsz);
        gemm(stream, core, wscr, heads, SB, 3 * Dsz, Dsz, Dsz, 1, Dsz, 3 * Dsz);
        cvt16(stream, heads, heads16, (long)SB * 3 * Dsz);
        // r_k = r @ r_w^T  [S x D] viewed [S,H,DH]
        cvt16(stream, Wr, wscr, (long)Dsz * Dsz);
        gemm(stream, rbuf, wscr, rk, Ssz, Dsz, Dsz, Dsz, 1, Dsz, Dsz);
        cvt16(stream, rk, rk16, (long)Ssz * Dsz);
        // q + rw_bias / q + rr_bias
        { long tot = (long)SB * Dsz;
          k_qbias<<<(tot + 255) / 256, 256, 0, stream>>>(heads, rw_b + (size_t)l * Dsz,
                                                         rr_b + (size_t)l * Dsz, qrw, qrr); }
        // AC[bt,i,j] = (q+rw) . k    batch bt=(b,h); head offset = b*3D + h*DH
        gemm(stream, qrw, heads16 + Dsz, scS, Ssz, Ssz, DHsz,
             (long)Bsz * Dsz, 1, (long)Bsz * 3 * Dsz, Ssz,
             Bsz * Hsz, Hsz,
             Dsz, DHsz, 3L * Dsz, DHsz, (long)Hsz * Ssz * Ssz, (long)Ssz * Ssz);
        // BDraw[bt,i,j] = (q+rr) . r_k
        gemm(stream, qrr, rk16, bdr, Ssz, Ssz, DHsz,
             (long)Bsz * Dsz, 1, Dsz, Ssz,
             Bsz * Hsz, Hsz,
             Dsz, DHsz, 0, DHsz, (long)Hsz * Ssz * Ssz, (long)Ssz * Ssz);
        { long tot = (long)Bsz * Hsz * Ssz * Ssz;
          k_shift_mask<<<(tot + 255) / 256, 256, 0, stream>>>(scS, bdr); }
        k_softmax<<<Bsz * Hsz * Ssz, 256, 0, stream>>>(scS, Ssz);
        // vec = prob @ v -> [S,B,D]   (B operand n-contiguous: BKC=0 path)
        gemm(stream, scS, heads16 + 2 * Dsz, vec, Ssz, DHsz, Ssz,
             Ssz, (long)Bsz * 3 * Dsz, 1, (long)Bsz * Dsz,
             Bsz * Hsz, Hsz,
             (long)Hsz * Ssz * Ssz, (long)Ssz * Ssz, 3L * Dsz, DHsz, Dsz, DHsz);
        // x = LN1(core + vec @ o_w^T)
        cvt16(stream, Wo, wscr, (long)Dsz * Dsz);
        gemm(stream, vec, wscr, xbuf, SB, Dsz, Dsz, Dsz, 1, Dsz, Dsz,
             1, 1, 0, 0, 0, 0, 0, 0, nullptr, core, Dsz);
        k_layernorm<<<SB, 256, 0, stream>>>(xbuf, ln1g + (size_t)l * Dsz, ln1b + (size_t)l * Dsz, xln);

        // Cross attention
        cvt16(stream, Wq, wscr, (long)Dsz * Dsz);
        gemm(stream, xln, wscr, q2, SB, Dsz, Dsz, Dsz, 1, Dsz, Dsz);
        cvt16(stream, Wkv, wscr, 2L * Dsz * Dsz);
        gemm(stream, enc_t, wscr, kv, Esz * Bsz, 2 * Dsz, Dsz, Dsz, 1, Dsz, 2 * Dsz);
        cvt16(stream, kv, kv16, (long)Esz * Bsz * 2 * Dsz);
        // score2 = SCALE * q2 . k2 + enc-mask
        gemm(stream, q2, kv16, sc2, Ssz, Esz, DHsz,
             (long)Bsz * Dsz, 1, (long)Bsz * 2 * Dsz, Esz,
             Bsz * Hsz, Hsz,
             Dsz, DHsz, 2L * Dsz, DHsz, (long)Hsz * Ssz * Esz, (long)Ssz * Esz,
             nullptr, nullptr, 0, input_mask, Esz, SCALEF, 0);
        k_softmax<<<Bsz * Hsz * Ssz, 256, 0, stream>>>(sc2, Esz);
        gemm(stream, sc2, kv16 + Dsz, vec2, Ssz, DHsz, Esz,
             Esz, (long)Bsz * 2 * Dsz, 1, (long)Bsz * Dsz,
             Bsz * Hsz, Hsz,
             (long)Hsz * Ssz * Esz, (long)Ssz * Esz, 2L * Dsz, DHsz, Dsz, DHsz);
        // y = LN2(x + vec2 @ io_w^T)
        cvt16(stream, Wio, wscr, (long)Dsz * Dsz);
        gemm(stream, vec2, wscr, ybuf, SB, Dsz, Dsz, Dsz, 1, Dsz, Dsz,
             1, 1, 0, 0, 0, 0, 0, 0, nullptr, xln, Dsz);
        k_layernorm<<<SB, 256, 0, stream>>>(ybuf, ln2g + (size_t)l * Dsz, ln2b + (size_t)l * Dsz, yln);

        // FFN with fused exact-GELU; second GEMM fuses bias + residual
        cvt16(stream, Wf1, wscr, (long)FFsz * Dsz);
        gemm(stream, yln, wscr, hff, SB, FFsz, Dsz, Dsz, 1, Dsz, FFsz,
             1, 1, 0, 0, 0, 0, 0, 0, f1b + (size_t)l * FFsz, nullptr, 0, nullptr, 0, 1.0f, 1);
        cvt16(stream, Wf2, wscr, (long)Dsz * FFsz);
        gemm(stream, hff, wscr, cbuf, SB, Dsz, FFsz, FFsz, 1, FFsz, Dsz,
             1, 1, 0, 0, 0, 0, 0, 0, f2b + (size_t)l * Dsz, yln, Dsz);
        k_layernorm<<<SB, 256, 0, stream>>>(cbuf, ln3g + (size_t)l * Dsz, ln3b + (size_t)l * Dsz, core);
    }

    // Heads: [S,B,D] -> [B,S,D]
    { long tot = (long)SB * Dsz;
      k_transpose<<<(tot + 255) / 256, 256, 0, stream>>>(core, cout, Ssz, Bsz); }
    cvt16(stream, out_w, wscr, (long)Dsz * Dsz);
    gemm(stream, cout, wscr, obuf, SB, Dsz, Dsz, Dsz, 1, Dsz, Dsz,
         1, 1, 0, 0, 0, 0, 0, 0, out_b);
    // vocab logits: 1024 x 32000 x 1024 (f16 word_emb: half the streaming bytes)
    gemm(stream, obuf, word16, vocab, SB, Vsz, Dsz, Dsz, 1, Dsz, Vsz);
    k_softmax<<<SB, 256, 0, stream>>>(vocab, Vsz);

    cvt16(stream, cp_w, wscr, (long)Dsz * Dsz);
    gemm(stream, cout, wscr, cph, SB, Dsz, Dsz, Dsz, 1, Dsz, Dsz,
         1, 1, 0, 0, 0, 0, 0, 0, cp_b);
    // input_logits[b,i,j] = copy_h[b,i,:] . encoder_rep[b,j,:] + mask
    gemm(stream, cph, enc16, ilog, Ssz, Esz, Dsz,
         Dsz, 1, Dsz, Esz,
         Bsz, 1,
         (long)Ssz * Dsz, 0, (long)Esz * Dsz, 0, (long)Ssz * Esz, 0,
         nullptr, nullptr, 0, input_mask, Esz, 1.0f, 0);
    k_softmax<<<SB, 256, 0, stream>>>(ilog, Esz);

    k_mode<<<SB, 256, 0, stream>>>(cout, md_w, md_b, modeb);
    k_mix<<<SB, 256, 0, stream>>>(vocab, modeb, ilog, input_ids);

    k_zero2<<<1, 32, 0, stream>>>(accum);
    k_loss<<<(SB + 255) / 256, 256, 0, stream>>>(vocab, decode_tgt, accum);
    k_finish<<<1, 32, 0, stream>>>(accum, (float*)d_out);
}
